// ImportanceWeightedAttention_49709951484792
// MI455X (gfx1250) — matrix operand order repaired and checked by reading, not compiled, not run on hardware
//
#include <hip/hip_runtime.h>

#define DM 1024
#define NH 16
#define HD 64
#define BB 8
#define MM 128
#define NN 8192
#define ARPB 64   // attention rows per block

typedef __bf16 bf16x16 __attribute__((ext_vector_type(16)));
typedef __bf16 bf16x8  __attribute__((ext_vector_type(8)));
typedef float  f32x8   __attribute__((ext_vector_type(8)));
typedef unsigned int u32x4 __attribute__((ext_vector_type(4)));
typedef int          i32x4 __attribute__((ext_vector_type(4)));
typedef int          i32x8 __attribute__((ext_vector_type(8)));

__device__ inline unsigned short f2bf(float f) {
  union { float f; unsigned u; } x; x.f = f;
  unsigned r = x.u + 0x7FFFu + ((x.u >> 16) & 1u);   // round-to-nearest-even
  return (unsigned short)(r >> 16);
}

__device__ inline f32x8 zero8() {
  f32x8 v;
#pragma unroll
  for (int i = 0; i < 8; ++i) v[i] = 0.f;
  return v;
}

__device__ inline f32x8 wmma_bf16(bf16x16 a, bf16x16 b, f32x8 c) {
  return __builtin_amdgcn_wmma_f32_16x16x32_bf16(false, a, false, b, (short)0, c, false, false);
}

__device__ inline unsigned lds_off(const void* p) { return (unsigned)(size_t)p; }

// TDM: load a (tile_d1 x tile_d0) bf16 tile, row stride `stride_e` elements, into LDS
// with +8-ushort row padding (pad_interval=32 DWORDs -> code 4, pad_amount=4 DWORDs -> code 3).
// Tensor dims == tile dims (tile is fully in-bounds). One scalar issue per tile (call from one wave).
__device__ inline void tdm_load_bf16_2d(unsigned lds, const unsigned short* gptr,
                                        unsigned tile_d0, unsigned tile_d1, unsigned stride_e) {
  unsigned long long ga = (unsigned long long)(size_t)gptr;
  u32x4 g0;
  g0[0] = 1u;                                                // count=1 (valid user D#)
  g0[1] = lds;                                               // lds_addr
  g0[2] = (unsigned)(ga & 0xFFFFFFFFu);                      // global_addr[31:0]
  g0[3] = (unsigned)((ga >> 32) & 0x1FFFFFFu) | (2u << 30);  // global_addr[56:32] | type=2
  i32x8 g1;
  g1[0] = (1 << 16) | (1 << 20) | (4 << 22) | (3 << 25);     // data_size=2B, pad_en, ivl=32dw, amt=4dw
  g1[1] = (int)((tile_d0 & 0xFFFFu) << 16);                  // tensor_dim0 lo16 (hi bits zero)
  g1[2] = (int)((tile_d1 & 0xFFFFu) << 16);                  // tensor_dim0 hi=0 | tensor_dim1 lo16
  g1[3] = (int)((tile_d0 & 0xFFFFu) << 16);                  // tensor_dim1 hi=0 | tile_dim0
  g1[4] = (int)(tile_d1 & 0xFFFFu);                          // tile_dim1 | tile_dim2=0
  g1[5] = (int)stride_e;                                     // tensor_dim0_stride lo32
  g1[6] = 0;                                                 // stride hi | dim1_stride lo
  g1[7] = 0;
  i32x4 z4; z4[0] = 0; z4[1] = 0; z4[2] = 0; z4[3] = 0;
  i32x8 z8;
#pragma unroll
  for (int i = 0; i < 8; ++i) z8[i] = 0;
  __builtin_amdgcn_tensor_load_to_lds(g0, g1, z4, z4, z8, 0);
}

// A-matrix fragment (16x32 bf16): element j -> k = kk + 8*half + (j&7) + 16*(j>>3)
__device__ inline bf16x16 ld_fragA(const unsigned short* base, int stride, int row, int kk) {
  const int half = (threadIdx.x >> 4) & 1;
  const unsigned short* p = base + row * stride + kk + half * 8;
  union { bf16x16 v; bf16x8 h[2]; } u;
  u.h[0] = *(const bf16x8*)(p);
  u.h[1] = *(const bf16x8*)(p + 16);
  return u.v;
}

// B-matrix fragment (32x16 bf16): element j -> k = kk + 8*half + (j&7) + 16*(j>>3), col = lane%16.
__device__ inline bf16x16 ld_fragB(const unsigned short* base, int stride, int row, int kk) {
  const int half = (threadIdx.x >> 4) & 1;
  const unsigned short* p = base + row * stride + kk + half * 8;
  union { bf16x16 v; bf16x8 h[2]; } u;
  u.h[0] = *(const bf16x8*)(p);
  u.h[1] = *(const bf16x8*)(p + 16);
  return u.v;
}

__device__ inline float hmax16(float v) {
#pragma unroll
  for (int m = 1; m < 16; m <<= 1) v = fmaxf(v, __shfl_xor(v, m, 32));
  return v;
}
__device__ inline float hsum16(float v) {
#pragma unroll
  for (int m = 1; m < 16; m <<= 1) v += __shfl_xor(v, m, 32);
  return v;
}

// ---------------- elementwise helpers ----------------
__global__ void cvt_bf16_kernel(const float* __restrict__ src, unsigned short* __restrict__ dst, int n) {
  int i = blockIdx.x * 256 + threadIdx.x;
  if (i < n) dst[i] = f2bf(src[i]);
}
__global__ void log_imp_kernel(const float* __restrict__ src, float* __restrict__ dst, int n) {
  int i = blockIdx.x * 256 + threadIdx.x;
  if (i < n) dst[i] = __logf(fmaxf(src[i], 1e-6f));
}

// ---------------- WMMA GEMM: out[r,c] = sum_k X[r,k] * W[c,k] + bias[c] ----------------
template <typename TIN, bool OUT_BF16>
__global__ __launch_bounds__(256) void gemm_wmma_kernel(
    const TIN* __restrict__ X, const unsigned short* __restrict__ Wbf,
    const float* __restrict__ bias, void* __restrict__ outp) {
  __shared__ unsigned short Xs[128][72];
  __shared__ unsigned short Ws[128][72];
  const int tid  = threadIdx.x;
  const int lane = tid & 31, ln = lane & 15, hi = (lane >> 4) & 1;
  const int w    = tid >> 5;
  const int wrow = (w >> 1) * 32;
  const int wcol = (w & 1) * 64;
  const size_t rblk = (size_t)blockIdx.y * 128;
  const int    nblk = blockIdx.x * 128;

  f32x8 acc[2][4];
#pragma unroll
  for (int mt = 0; mt < 2; ++mt)
#pragma unroll
    for (int nt = 0; nt < 4; ++nt) acc[mt][nt] = zero8();

  for (int kt = 0; kt < 16; ++kt) {
    const int k0 = kt * 64;
    __syncthreads();
    // W tile 128x64 via Tensor Data Mover (one scalar issue from wave 0)
    if (w == 0)
      tdm_load_bf16_2d(lds_off(&Ws[0][0]), &Wbf[(size_t)nblk * DM + k0], 64, 128, DM);
    // X tile 128x64 (manual: needs fp32->bf16 conversion on one path)
    if constexpr (sizeof(TIN) == 4) {
#pragma unroll
      for (int it = 0; it < 8; ++it) {
        int u = tid + 256 * it;
        int r = u >> 4, c4 = (u & 15) * 4;
        float4 f = *(const float4*)&X[(rblk + r) * DM + k0 + c4];
        Xs[r][c4 + 0] = f2bf(f.x);
        Xs[r][c4 + 1] = f2bf(f.y);
        Xs[r][c4 + 2] = f2bf(f.z);
        Xs[r][c4 + 3] = f2bf(f.w);
      }
    } else {
#pragma unroll
      for (int it = 0; it < 4; ++it) {
        int u = tid + 256 * it;
        int r = u >> 3, c8 = (u & 7) * 8;
        *(uint4*)&Xs[r][c8] = *(const uint4*)&X[(rblk + r) * DM + k0 + c8];
      }
    }
    if (kt + 1 < 16)
      __builtin_prefetch(&X[(rblk + (tid & 127)) * DM + k0 + 64], 0, 1);
    if (w == 0) __builtin_amdgcn_s_wait_tensorcnt(0);
    __syncthreads();
#pragma unroll
    for (int kk = 0; kk < 64; kk += 32) {
      bf16x16 af[2], bfr[4];
#pragma unroll
      for (int mt = 0; mt < 2; ++mt) af[mt] = ld_fragA(&Xs[0][0], 72, wrow + mt * 16 + ln, kk);
#pragma unroll
      for (int nt = 0; nt < 4; ++nt) bfr[nt] = ld_fragB(&Ws[0][0], 72, wcol + nt * 16 + ln, kk);
#pragma unroll
      for (int mt = 0; mt < 2; ++mt)
#pragma unroll
        for (int nt = 0; nt < 4; ++nt) acc[mt][nt] = wmma_bf16(af[mt], bfr[nt], acc[mt][nt]);
    }
  }
#pragma unroll
  for (int mt = 0; mt < 2; ++mt)
#pragma unroll
    for (int nt = 0; nt < 4; ++nt)
#pragma unroll
      for (int i = 0; i < 8; ++i) {
        size_t row = rblk + wrow + mt * 16 + i + 8 * hi;
        int col = nblk + wcol + nt * 16 + ln;
        float v = acc[mt][nt][i] + bias[col];
        if constexpr (OUT_BF16) ((unsigned short*)outp)[row * DM + col] = f2bf(v);
        else                    ((float*)outp)[row * DM + col] = v;
      }
}

// ---------------- flash attention: 2 blocks of 64 rows per (b,h) ----------------
__global__ __launch_bounds__(128) void attn_kernel(
    const unsigned short* __restrict__ Qbf, const unsigned short* __restrict__ Kbf,
    const unsigned short* __restrict__ Vbf, const float* __restrict__ logb,
    unsigned short* __restrict__ Obf) {
  __shared__ unsigned short Qs[ARPB][72];
  __shared__ unsigned short Ks[64][72];
  __shared__ unsigned short Vt[64][72];     // transposed V: Vt[c][keyrow]
  __shared__ unsigned short Ps[4][16][72];  // per-wave P tile
  __shared__ float biasS[64];

  const int tid = threadIdx.x;
  const int w = tid >> 5, lane = tid & 31, ln = lane & 15, hi = (lane >> 4) & 1;
  const int bx = blockIdx.x;
  const int b = bx / (NH * 2);
  const int rem = bx % (NH * 2);
  const int h = rem >> 1;
  const int mb = (rem & 1) * ARPB;
  const int r0 = w * 16;

  // Q tile 64x64 via TDM (waited on inside the first chunk iteration)
  if (w == 0)
    tdm_load_bf16_2d(lds_off(&Qs[0][0]), &Qbf[((size_t)(b * MM + mb)) * DM + h * HD], 64, ARPB, DM);

  f32x8 acc_o[4];
  float m_i[8], l_i[8];
#pragma unroll
  for (int t = 0; t < 4; ++t) acc_o[t] = zero8();
#pragma unroll
  for (int i = 0; i < 8; ++i) { m_i[i] = -1e30f; l_i[i] = 0.f; }

  for (int n0 = 0; n0 < NN; n0 += 64) {
    __syncthreads();
    // K chunk 64x64 via TDM
    if (w == 0)
      tdm_load_bf16_2d(lds_off(&Ks[0][0]), &Kbf[((size_t)(b * NN + n0)) * DM + h * HD], 64, 64, DM);
    // V chunk, transposed into Vt (manual)
#pragma unroll
    for (int it = 0; it < 4; ++it) {
      int u = tid + 128 * it;            // 512 units of 8 ushorts
      int r = u >> 3, c8 = (u & 7) * 8;
      unsigned short tmp[8] __attribute__((aligned(16)));
      *(uint4*)tmp = *(const uint4*)&Vbf[((size_t)(b * NN + n0 + r)) * DM + h * HD + c8];
#pragma unroll
      for (int j = 0; j < 8; ++j) Vt[c8 + j][r] = tmp[j];
    }
    if (tid < 64) biasS[tid] = logb[(size_t)b * NN + n0 + tid];
    if (w == 0) __builtin_amdgcn_s_wait_tensorcnt(0);
    __syncthreads();

    // S = Q K^T  (16 query rows x 64 keys per wave)
    f32x8 acc_s[4];
#pragma unroll
    for (int t = 0; t < 4; ++t) acc_s[t] = zero8();
#pragma unroll
    for (int kk = 0; kk < HD; kk += 32) {
      bf16x16 af = ld_fragA(&Qs[0][0], 72, r0 + ln, kk);
#pragma unroll
      for (int t = 0; t < 4; ++t) {
        bf16x16 bfr = ld_fragB(&Ks[0][0], 72, t * 16 + ln, kk);
        acc_s[t] = wmma_bf16(af, bfr, acc_s[t]);
      }
    }

    // online softmax (each row lives in a 16-lane group per C layout)
    float bc[4];
#pragma unroll
    for (int t = 0; t < 4; ++t) bc[t] = biasS[t * 16 + ln];
#pragma unroll
    for (int i = 0; i < 8; ++i) {
      float v0 = acc_s[0][i] * 0.125f + bc[0];
      float v1 = acc_s[1][i] * 0.125f + bc[1];
      float v2 = acc_s[2][i] * 0.125f + bc[2];
      float v3 = acc_s[3][i] * 0.125f + bc[3];
      float mx = hmax16(fmaxf(fmaxf(v0, v1), fmaxf(v2, v3)));
      float mnew = fmaxf(m_i[i], mx);
      float alpha = __expf(m_i[i] - mnew);
      float p0 = __expf(v0 - mnew), p1 = __expf(v1 - mnew);
      float p2 = __expf(v2 - mnew), p3 = __expf(v3 - mnew);
      float rs = hsum16(p0 + p1 + p2 + p3);
      l_i[i] = l_i[i] * alpha + rs;
      m_i[i] = mnew;
      int ri = i + 8 * hi;
      Ps[w][ri][ 0 + ln] = f2bf(p0);
      Ps[w][ri][16 + ln] = f2bf(p1);
      Ps[w][ri][32 + ln] = f2bf(p2);
      Ps[w][ri][48 + ln] = f2bf(p3);
#pragma unroll
      for (int t = 0; t < 4; ++t) acc_o[t][i] *= alpha;
    }
    __builtin_amdgcn_wave_barrier();   // per-wave LDS tile: DS ops are in-order

    // O += P V
#pragma unroll
    for (int kk = 0; kk < 64; kk += 32) {
      bf16x16 af = ld_fragA(&Ps[w][0][0], 72, ln, kk);
#pragma unroll
      for (int t = 0; t < 4; ++t) {
        bf16x16 bfr = ld_fragB(&Vt[0][0], 72, t * 16 + ln, kk);
        acc_o[t] = wmma_bf16(af, bfr, acc_o[t]);
      }
    }
  }

#pragma unroll
  for (int i = 0; i < 8; ++i) {
    float inv = 1.f / l_i[i];
    int m = mb + r0 + i + 8 * hi;
#pragma unroll
    for (int t = 0; t < 4; ++t)
      Obf[((size_t)(b * MM + m)) * DM + h * HD + t * 16 + ln] = f2bf(acc_o[t][i] * inv);
  }
}

extern "C" void kernel_launch(void* const* d_in, const int* in_sizes, int n_in,
                              void* d_out, int out_size, void* d_ws, size_t ws_size,
                              hipStream_t stream) {
  (void)in_sizes; (void)n_in; (void)out_size; (void)ws_size;
  const float* query = (const float*)d_in[0];
  const float* key_  = (const float*)d_in[1];
  const float* value = (const float*)d_in[2];
  const float* imp   = (const float*)d_in[3];
  const float* Wq = (const float*)d_in[4];
  const float* bq = (const float*)d_in[5];
  const float* Wk = (const float*)d_in[6];
  const float* bk = (const float*)d_in[7];
  const float* Wv = (const float*)d_in[8];
  const float* bv = (const float*)d_in[9];
  const float* Wo = (const float*)d_in[10];
  const float* bo = (const float*)d_in[11];
  float* out = (float*)d_out;

  char* ws = (char*)d_ws;
  size_t off = 0;
  auto take = [&](size_t bytes) {
    char* p = ws + off;
    off += (bytes + 255) & ~(size_t)255;
    return p;
  };
  unsigned short* wqbf = (unsigned short*)take((size_t)DM * DM * 2);
  unsigned short* wkbf = (unsigned short*)take((size_t)DM * DM * 2);
  unsigned short* wvbf = (unsigned short*)take((size_t)DM * DM * 2);
  unsigned short* wobf = (unsigned short*)take((size_t)DM * DM * 2);
  float*          logb = (float*)take((size_t)BB * NN * 4);
  unsigned short* qbf  = (unsigned short*)take((size_t)BB * MM * DM * 2);
  unsigned short* kbf  = (unsigned short*)take((size_t)BB * NN * DM * 2);
  unsigned short* vbf  = (unsigned short*)take((size_t)BB * NN * DM * 2);
  unsigned short* obf  = (unsigned short*)take((size_t)BB * MM * DM * 2);

  const int nW = DM * DM;
  cvt_bf16_kernel<<<(nW + 255) / 256, 256, 0, stream>>>(Wq, wqbf, nW);
  cvt_bf16_kernel<<<(nW + 255) / 256, 256, 0, stream>>>(Wk, wkbf, nW);
  cvt_bf16_kernel<<<(nW + 255) / 256, 256, 0, stream>>>(Wv, wvbf, nW);
  cvt_bf16_kernel<<<(nW + 255) / 256, 256, 0, stream>>>(Wo, wobf, nW);
  log_imp_kernel<<<(BB * NN + 255) / 256, 256, 0, stream>>>(imp, logb, BB * NN);

  gemm_wmma_kernel<float, true><<<dim3(DM / 128, (BB * MM) / 128), 256, 0, stream>>>(query, wqbf, bq, qbf);
  gemm_wmma_kernel<float, true><<<dim3(DM / 128, (BB * NN) / 128), 256, 0, stream>>>(key_, wkbf, bk, kbf);
  gemm_wmma_kernel<float, true><<<dim3(DM / 128, (BB * NN) / 128), 256, 0, stream>>>(value, wvbf, bv, vbf);

  attn_kernel<<<dim3(BB * NH * 2), 128, 0, stream>>>(qbf, kbf, vbf, logb, obf);

  gemm_wmma_kernel<unsigned short, false><<<dim3(DM / 128, (BB * MM) / 128), 256, 0, stream>>>(obf, wobf, bo, out);
}
